// DEQTransformerLM_28956669509878
// MI455X (gfx1250) — matrix operand order repaired and checked by reading, not compiled, not run on hardware
//
#include <hip/hip_runtime.h>

// ---------------- problem constants (from reference) ----------------
#define BSZ    16
#define DMODEL 1024
#define QLEN   256
#define MLEN   256
#define KLEN   512      // qlen + mlen
#define NHEAD  16
#define DHEAD  64
#define DINNER 4096
#define HD3    3072     // 3 * n_head * d_head
#define LN_EPS 1e-5f
#define ATT_SCALE 0.125f // 1/sqrt(64)

typedef __attribute__((ext_vector_type(16))) __bf16       v16bf;
typedef __attribute__((ext_vector_type(8)))  float        v8f;
typedef __attribute__((ext_vector_type(4)))  unsigned int u32x4;

union FragAB { v16bf v; u32x4 u[2]; };

__device__ __forceinline__ __bf16 f2bf(float x) {
  unsigned u = __builtin_bit_cast(unsigned, x);
  u += 0x7fffu + ((u >> 16) & 1u);            // round-to-nearest-even
  unsigned short s = (unsigned short)(u >> 16);
  return __builtin_bit_cast(__bf16, s);
}

// ---------------- epilogue flags ----------------
#define EPI_ADDMAT 1
#define EPI_BIASM  2
#define EPI_BIASN  4
#define EPI_RELU   8
#define EPI_BF16   16

// =====================================================================
// Batched NT GEMM:  C[z][m][n] = sum_k A[z][m][k] * B[z][n][k]  (bf16 in,
// f32 accum via v_wmma_f32_16x16x32_bf16).
// Register-blocked: each wave owns a 32(M) x 64(N) macro-tile
//   -> 2 A frags x 4 B frags -> 8 WMMAs per K-step (64 acc VGPRs/lane).
// Block = 128 thr = 4 waves stacked along M -> block tile 128 x 64.
// Per-batch operand offsets: (z/div)*s1 + (z%div)*s2.
// Requires M % 128 == 0 (or M rounded by grid with guard), N % 64 == 0,
// K % 32 == 0 -- true for every call in this layer.
// =====================================================================
__global__ __launch_bounds__(128)
void gemm_bf16_wmma(const __bf16* __restrict__ A, const __bf16* __restrict__ B,
                    float* __restrict__ C, const float* __restrict__ addend,
                    const float* __restrict__ bias, __bf16* __restrict__ Cbf,
                    int M, int N, int K, int lda, int ldb, int ldc,
                    long aS1, long aS2, int aDiv,
                    long bS1, long bS2, int bDiv,
                    long cS1, long cS2, int cDiv,
                    long dS1, int epi)
{
  const int z = blockIdx.z;
  const long aOff = (long)(z / aDiv) * aS1 + (long)(z % aDiv) * aS2;
  const long bOff = (long)(z / bDiv) * bS1 + (long)(z % bDiv) * bS2;
  const long cOff = (long)(z / cDiv) * cS1 + (long)(z % cDiv) * cS2;
  const long dOff = (long)z * dS1;

  const int wave = threadIdx.x >> 5;
  const int lane = threadIdx.x & 31;
  const int rowBase = blockIdx.y * 128 + wave * 32;   // 32 M-rows per wave
  const int colBase = blockIdx.x * 64;                // 64 N-cols per wave
  if (rowBase >= M || colBase >= N) return;

  const int r    = lane & 15;   // M row (A) / N col (B) owned by this lane
  const int hsel = lane >> 4;   // half-wave select
  const int ka0  = hsel * 8;    // A: lanes 0-15 K {0..7,16..23}; 16-31 {8..15,24..31}
  const int kb0  = hsel * 8;    // B: lanes 0-15 K {0..7,16..23}; 16-31 {8..15,24..31}

  const __bf16* Ap0 = A + aOff + (long)(rowBase + r) * lda;
  const __bf16* Ap1 = Ap0 + (long)16 * lda;
  const __bf16* Bp0 = B + bOff + (long)(colBase + r) * ldb;
  const __bf16* Bp1 = Bp0 + (long)16 * ldb;
  const __bf16* Bp2 = Bp0 + (long)32 * ldb;
  const __bf16* Bp3 = Bp0 + (long)48 * ldb;

  v8f acc[2][4];
#pragma unroll
  for (int ti = 0; ti < 2; ++ti)
#pragma unroll
    for (int tj = 0; tj < 4; ++tj)
      acc[ti][tj] = (v8f){0.f, 0.f, 0.f, 0.f, 0.f, 0.f, 0.f, 0.f};

  FragAB fa[2], fb[4];
  for (int kk = 0; kk < K; kk += 32) {
    fa[0].u[0] = *reinterpret_cast<const u32x4*>(Ap0 + kk + ka0);
    fa[0].u[1] = *reinterpret_cast<const u32x4*>(Ap0 + kk + ka0 + 16);
    fa[1].u[0] = *reinterpret_cast<const u32x4*>(Ap1 + kk + ka0);
    fa[1].u[1] = *reinterpret_cast<const u32x4*>(Ap1 + kk + ka0 + 16);
    fb[0].u[0] = *reinterpret_cast<const u32x4*>(Bp0 + kk + kb0);
    fb[0].u[1] = *reinterpret_cast<const u32x4*>(Bp0 + kk + kb0 + 16);
    fb[1].u[0] = *reinterpret_cast<const u32x4*>(Bp1 + kk + kb0);
    fb[1].u[1] = *reinterpret_cast<const u32x4*>(Bp1 + kk + kb0 + 16);
    fb[2].u[0] = *reinterpret_cast<const u32x4*>(Bp2 + kk + kb0);
    fb[2].u[1] = *reinterpret_cast<const u32x4*>(Bp2 + kk + kb0 + 16);
    fb[3].u[0] = *reinterpret_cast<const u32x4*>(Bp3 + kk + kb0);
    fb[3].u[1] = *reinterpret_cast<const u32x4*>(Bp3 + kk + kb0 + 16);

    if (kk + 32 < K) {  // pull next K-step lines toward the WGP (global_prefetch_b8)
      __builtin_prefetch((const void*)(Ap0 + kk + 32 + ka0), 0, 1);
      __builtin_prefetch((const void*)(Bp0 + kk + 32 + kb0), 0, 1);
    }

#pragma unroll
    for (int ti = 0; ti < 2; ++ti)
#pragma unroll
      for (int tj = 0; tj < 4; ++tj)
        acc[ti][tj] = __builtin_amdgcn_wmma_f32_16x16x32_bf16(
            false, fa[ti].v, false, fb[tj].v, (short)0, acc[ti][tj], false, false);
  }

  // C layout: VGPR v -> M = v + 8*hsel ; N = lane&15
#pragma unroll
  for (int ti = 0; ti < 2; ++ti) {
#pragma unroll
    for (int tj = 0; tj < 4; ++tj) {
      const int gn    = colBase + tj * 16 + r;
      const int mBase = rowBase + ti * 16 + hsel * 8;
#pragma unroll
      for (int vv = 0; vv < 8; ++vv) {
        const int gm = mBase + vv;
        const long idx = cOff + (long)gm * ldc + gn;
        float x = acc[ti][tj][vv];
        if (epi & EPI_ADDMAT) x += addend[dOff + (long)gm * ldc + gn];
        if (epi & EPI_BIASM)  x += bias[gm];
        if (epi & EPI_BIASN)  x += bias[gn];
        if (epi & EPI_RELU)   x = fmaxf(x, 0.f);
        C[idx] = x;
        if (epi & EPI_BF16)   Cbf[idx] = f2bf(x);
      }
    }
  }
}

// =====================================================================
// Elementwise prep kernels
// =====================================================================
__global__ void conv_f32_bf16(const float* __restrict__ in, __bf16* __restrict__ out, int n) {
  int i = blockIdx.x * blockDim.x + threadIdx.x;
  if (i < n) out[i] = f2bf(in[i]);
}

// catT[b][kl][d] = (kl < MLEN) ? mems[b][d][kl] : z1ss[b][d][kl-MLEN]
__global__ void build_catT(const float* __restrict__ mems, const float* __restrict__ z1ss,
                           __bf16* __restrict__ catT) {
  int i = blockIdx.x * blockDim.x + threadIdx.x;
  if (i >= BSZ * KLEN * DMODEL) return;
  int d  = i % DMODEL;
  int kl = (i / DMODEL) % KLEN;
  int b  = i / (DMODEL * KLEN);
  float v = (kl < MLEN) ? mems[((long)b * DMODEL + d) * MLEN + kl]
                        : z1ss[((long)b * DMODEL + d) * QLEN + (kl - MLEN)];
  catT[i] = f2bf(v);
}

// posT[j][d] = pos_emb[d][j]
__global__ void build_posT(const float* __restrict__ pos, __bf16* __restrict__ posT) {
  int i = blockIdx.x * blockDim.x + threadIdx.x;
  if (i >= KLEN * DMODEL) return;
  int d = i % DMODEL;
  int j = i / DMODEL;
  posT[i] = f2bf(pos[(long)d * KLEN + j]);
}

// Aq[(b*16+n)][i][d] = bf16(wheads[b][n*64+d][256+i] + r_w_bias[n][d]); Aqr with r_r_bias
__global__ void prep_q(const float* __restrict__ wheads, const float* __restrict__ rwb,
                       const float* __restrict__ rrb,
                       __bf16* __restrict__ Aq, __bf16* __restrict__ Aqr) {
  int i = blockIdx.x * blockDim.x + threadIdx.x;
  if (i >= BSZ * NHEAD * QLEN * DHEAD) return;
  int d  = i % DHEAD;
  int qi = (i / DHEAD) % QLEN;
  int nh = (i / (DHEAD * QLEN)) % NHEAD;
  int b  = i / (DHEAD * QLEN * NHEAD);
  float w = wheads[((long)b * HD3 + nh * DHEAD + d) * KLEN + MLEN + qi];
  Aq[i]  = f2bf(w + rwb[nh * DHEAD + d]);
  Aqr[i] = f2bf(w + rrb[nh * DHEAD + d]);
}

// Kt[(b*16+n)][j][d] = bf16(wheads[b][1024 + n*64+d][j])
__global__ void prep_kt(const float* __restrict__ wheads, __bf16* __restrict__ Kt) {
  int i = blockIdx.x * blockDim.x + threadIdx.x;
  if (i >= BSZ * NHEAD * KLEN * DHEAD) return;
  int d  = i % DHEAD;
  int j  = (i / DHEAD) % KLEN;
  int nh = (i / (DHEAD * KLEN)) % NHEAD;
  int b  = i / (DHEAD * KLEN * NHEAD);
  Kt[i] = f2bf(wheads[((long)b * HD3 + DMODEL + nh * DHEAD + d) * KLEN + j]);
}

// Vb[b][c][j] = bf16(wheads[b][2048 + c][j])   (c = n*64+d, row-major over j)
__global__ void prep_v(const float* __restrict__ wheads, __bf16* __restrict__ Vb) {
  int i = blockIdx.x * blockDim.x + threadIdx.x;
  if (i >= BSZ * DMODEL * KLEN) return;
  int j = i % KLEN;
  int c = (i / KLEN) % DMODEL;
  int b = i / (KLEN * DMODEL);
  Vb[i] = f2bf(wheads[((long)b * HD3 + 2 * DMODEL + c) * KLEN + j]);
}

// Rt[n][j][d] = bf16(rheadk[n*64+d][j])
__global__ void prep_rt(const float* __restrict__ rheadk, __bf16* __restrict__ Rt) {
  int i = blockIdx.x * blockDim.x + threadIdx.x;
  if (i >= NHEAD * KLEN * DHEAD) return;
  int d  = i % DHEAD;
  int j  = (i / DHEAD) % KLEN;
  int nh = i / (DHEAD * KLEN);
  Rt[i] = f2bf(rheadk[((long)nh * DHEAD + d) * KLEN + j]);
}

// =====================================================================
// Fused rel-shift + mask + softmax.  One block per (b,n,i) row of 512.
// rel-shift: out[i][j] = (jp>0) ? BD[ip][jp-1] : 0 with f=i*512+j+256,
// ip=f/513, jp=f%513.  Visible window: i+128 < j <= i+256.
// =====================================================================
__global__ __launch_bounds__(256)
void softmax_kernel(const float* __restrict__ AC, const float* __restrict__ BD,
                    __bf16* __restrict__ prob) {
  const int i  = blockIdx.x & 255;
  const int bn = blockIdx.x >> 8;
  const float* ac = AC + ((long)bn * QLEN + i) * KLEN;
  const float* bd = BD + (long)bn * QLEN * KLEN;
  const int t = threadIdx.x;

  float s[2];
#pragma unroll
  for (int e = 0; e < 2; ++e) {
    int j = t + e * 256;
    float v = -1e30f;
    if (j > i + MLEN - 128 && j <= i + MLEN) {   // inside local causal window
      int f  = i * KLEN + j + QLEN;
      int ip = f / (KLEN + 1);
      int jp = f % (KLEN + 1);
      float bdv = (jp > 0) ? bd[(long)ip * KLEN + (jp - 1)] : 0.f;
      v = (ac[j] + bdv) * ATT_SCALE;
    }
    s[e] = v;
  }

  __shared__ float red[256];
  red[t] = fmaxf(s[0], s[1]);
  __syncthreads();
  for (int off = 128; off > 0; off >>= 1) {
    if (t < off) red[t] = fmaxf(red[t], red[t + off]);
    __syncthreads();
  }
  float mx = red[0];
  __syncthreads();

  float e0 = __expf(s[0] - mx);
  float e1 = __expf(s[1] - mx);
  red[t] = e0 + e1;
  __syncthreads();
  for (int off = 128; off > 0; off >>= 1) {
    if (t < off) red[t] += red[t + off];
    __syncthreads();
  }
  float inv = 1.f / red[0];

  long pb = ((long)bn * QLEN + i) * KLEN;
  prob[pb + t]       = f2bf(e0 * inv);
  prob[pb + t + 256] = f2bf(e1 * inv);
}

// =====================================================================
// LayerNorm 1: h[b][i][:] = LN_d(attn_out[b][:][i] + z1ss[b][:][i])
// =====================================================================
__global__ __launch_bounds__(256)
void ln_h_kernel(const float* __restrict__ attn_out, const float* __restrict__ z1ss,
                 float* __restrict__ h, __bf16* __restrict__ hbf) {
  const int b = blockIdx.x >> 8;
  const int i = blockIdx.x & 255;
  const int t = threadIdx.x;
  const long baseIn = (long)b * DMODEL * QLEN + i;
  float x[4], sum = 0.f, sq = 0.f;
#pragma unroll
  for (int e = 0; e < 4; ++e) {
    int d = t + e * 256;
    float v = attn_out[baseIn + (long)d * QLEN] + z1ss[baseIn + (long)d * QLEN];
    x[e] = v; sum += v; sq += v * v;
  }
  __shared__ float rs[256], rq[256];
  rs[t] = sum; rq[t] = sq;
  __syncthreads();
  for (int off = 128; off > 0; off >>= 1) {
    if (t < off) { rs[t] += rs[t + off]; rq[t] += rq[t + off]; }
    __syncthreads();
  }
  float mean = rs[0] * (1.f / DMODEL);
  float var  = rq[0] * (1.f / DMODEL) - mean * mean;
  float rstd = rsqrtf(var + LN_EPS);
  long baseOut = ((long)b * QLEN + i) * DMODEL;
#pragma unroll
  for (int e = 0; e < 4; ++e) {
    int d = t + e * 256;
    float v = (x[e] - mean) * rstd;
    h[baseOut + d]   = v;
    hbf[baseOut + d] = f2bf(v);
  }
}

// LayerNorm 2: out[b][d][i] = LN_d(ff2[b][i][:] + h[b][i][:]) (writes transposed)
__global__ __launch_bounds__(256)
void ln_out_kernel(const float* __restrict__ ff2, const float* __restrict__ h,
                   float* __restrict__ out) {
  const int b = blockIdx.x >> 8;
  const int i = blockIdx.x & 255;
  const int t = threadIdx.x;
  const long baseIn = ((long)b * QLEN + i) * DMODEL;
  float x[4], sum = 0.f, sq = 0.f;
#pragma unroll
  for (int e = 0; e < 4; ++e) {
    int d = t + e * 256;
    float v = ff2[baseIn + d] + h[baseIn + d];
    x[e] = v; sum += v; sq += v * v;
  }
  __shared__ float rs[256], rq[256];
  rs[t] = sum; rq[t] = sq;
  __syncthreads();
  for (int off = 128; off > 0; off >>= 1) {
    if (t < off) { rs[t] += rs[t + off]; rq[t] += rq[t + off]; }
    __syncthreads();
  }
  float mean = rs[0] * (1.f / DMODEL);
  float var  = rq[0] * (1.f / DMODEL) - mean * mean;
  float rstd = rsqrtf(var + LN_EPS);
#pragma unroll
  for (int e = 0; e < 4; ++e) {
    int d = t + e * 256;
    out[((long)b * DMODEL + d) * QLEN + i] = (x[e] - mean) * rstd;
  }
}

// =====================================================================
// host side
// =====================================================================
static inline void gemmL(hipStream_t st, const __bf16* A, const __bf16* B, float* C,
                         const float* add, const float* bias, __bf16* Cbf,
                         int M, int N, int K, int lda, int ldb, int ldc,
                         long aS1, long aS2, int aDiv,
                         long bS1, long bS2, int bDiv,
                         long cS1, long cS2, int cDiv,
                         long dS1, int epi, int batches) {
  dim3 grid((N + 63) / 64, (M + 127) / 128, batches);
  gemm_bf16_wmma<<<grid, 128, 0, st>>>(A, B, C, add, bias, Cbf, M, N, K, lda, ldb, ldc,
                                       aS1, aS2, aDiv, bS1, bS2, bDiv, cS1, cS2, cDiv,
                                       dS1, epi);
}

static inline int G(long n) { return (int)((n + 255) / 256); }

extern "C" void kernel_launch(void* const* d_in, const int* in_sizes, int n_in,
                              void* d_out, int out_size, void* d_ws, size_t ws_size,
                              hipStream_t stream) {
  (void)in_sizes; (void)n_in; (void)out_size; (void)ws_size;
  const float* z1ss = (const float*)d_in[0];
  const float* uss  = (const float*)d_in[1];
  const float* mems = (const float*)d_in[2];
  const float* pos  = (const float*)d_in[3];
  const float* Wqkv = (const float*)d_in[4];
  const float* Wr   = (const float*)d_in[5];
  const float* Wo   = (const float*)d_in[6];
  const float* bo   = (const float*)d_in[7];
  const float* rwb  = (const float*)d_in[8];
  const float* rrb  = (const float*)d_in[9];
  const float* W1   = (const float*)d_in[10];
  const float* b1   = (const float*)d_in[11];
  const float* W2   = (const float*)d_in[12];
  const float* b2   = (const float*)d_in[13];
  float* out = (float*)d_out;

  // ---- workspace carve-out ----
  char* ws = (char*)d_ws;
  size_t cur = 0;
  auto alloc = [&](size_t bytes) -> void* {
    void* p = ws + cur;
    cur = (cur + bytes + 255) & ~(size_t)255;
    return p;
  };
  __bf16* catT   = (__bf16*)alloc((size_t)BSZ * KLEN * DMODEL * 2);
  __bf16* posT   = (__bf16*)alloc((size_t)KLEN * DMODEL * 2);
  __bf16* WqkvB  = (__bf16*)alloc((size_t)HD3 * DMODEL * 2);
  __bf16* WrB    = (__bf16*)alloc((size_t)DMODEL * DMODEL * 2);
  __bf16* WoB    = (__bf16*)alloc((size_t)DMODEL * DMODEL * 2);
  __bf16* W1B    = (__bf16*)alloc((size_t)DINNER * DMODEL * 2);
  __bf16* W2B    = (__bf16*)alloc((size_t)DMODEL * DINNER * 2);
  float*  wheads = (float*)alloc((size_t)BSZ * HD3 * KLEN * 4);
  float*  rheadk = (float*)alloc((size_t)DMODEL * KLEN * 4);
  __bf16* Aq     = (__bf16*)alloc((size_t)BSZ * NHEAD * QLEN * DHEAD * 2);
  __bf16* Aqr    = (__bf16*)alloc((size_t)BSZ * NHEAD * QLEN * DHEAD * 2);
  __bf16* Kt     = (__bf16*)alloc((size_t)BSZ * NHEAD * KLEN * DHEAD * 2);
  __bf16* Vb     = (__bf16*)alloc((size_t)BSZ * DMODEL * KLEN * 2);
  __bf16* Rt     = (__bf16*)alloc((size_t)NHEAD * KLEN * DHEAD * 2);
  float*  ACb    = (float*)alloc((size_t)BSZ * NHEAD * QLEN * KLEN * 4);
  float*  BDb    = (float*)alloc((size_t)BSZ * NHEAD * QLEN * KLEN * 4);
  __bf16* prob   = (__bf16*)alloc((size_t)BSZ * NHEAD * QLEN * KLEN * 2);
  float*  avF    = (float*)alloc((size_t)BSZ * QLEN * DMODEL * 4);
  __bf16* avB    = (__bf16*)alloc((size_t)BSZ * QLEN * DMODEL * 2);
  float*  attnO  = (float*)alloc((size_t)BSZ * DMODEL * QLEN * 4);
  float*  hF     = (float*)alloc((size_t)BSZ * QLEN * DMODEL * 4);
  __bf16* hB     = (__bf16*)alloc((size_t)BSZ * QLEN * DMODEL * 2);
  float*  ff1F   = (float*)alloc((size_t)BSZ * QLEN * DINNER * 4);
  __bf16* ff1B   = (__bf16*)alloc((size_t)BSZ * QLEN * DINNER * 2);
  float*  ff2F   = (float*)alloc((size_t)BSZ * QLEN * DMODEL * 4);

  // ---- 1. bf16 conversions / transposes ----
  conv_f32_bf16<<<G((long)HD3 * DMODEL),   256, 0, stream>>>(Wqkv, WqkvB, HD3 * DMODEL);
  conv_f32_bf16<<<G((long)DMODEL * DMODEL),256, 0, stream>>>(Wr,   WrB,   DMODEL * DMODEL);
  conv_f32_bf16<<<G((long)DMODEL * DMODEL),256, 0, stream>>>(Wo,   WoB,   DMODEL * DMODEL);
  conv_f32_bf16<<<G((long)DINNER * DMODEL),256, 0, stream>>>(W1,   W1B,   DINNER * DMODEL);
  conv_f32_bf16<<<G((long)DMODEL * DINNER),256, 0, stream>>>(W2,   W2B,   DMODEL * DINNER);
  build_catT<<<G((long)BSZ * KLEN * DMODEL), 256, 0, stream>>>(mems, z1ss, catT);
  build_posT<<<G((long)KLEN * DMODEL),       256, 0, stream>>>(pos, posT);

  // ---- 2. w_heads = Wqkv @ cat + uss : per b, M=3072 N=512 K=1024 ----
  gemmL(stream, WqkvB, catT, wheads, uss, nullptr, nullptr,
        HD3, KLEN, DMODEL, DMODEL, DMODEL, KLEN,
        0, 0, 1, (long)KLEN * DMODEL, 0, 1, (long)HD3 * KLEN, 0, 1,
        (long)HD3 * KLEN, EPI_ADDMAT, BSZ);

  // ---- 3. r_head_k = Wr @ pos_emb : M=1024 N=512 K=1024 ----
  gemmL(stream, WrB, posT, rheadk, nullptr, nullptr, nullptr,
        DMODEL, KLEN, DMODEL, DMODEL, DMODEL, KLEN,
        0, 0, 1, 0, 0, 1, 0, 0, 1, 0, 0, 1);

  // ---- 4. build attention operands (bf16) ----
  prep_q <<<G((long)BSZ * NHEAD * QLEN * DHEAD), 256, 0, stream>>>(wheads, rwb, rrb, Aq, Aqr);
  prep_kt<<<G((long)BSZ * NHEAD * KLEN * DHEAD), 256, 0, stream>>>(wheads, Kt);
  prep_v <<<G((long)BSZ * DMODEL * KLEN),        256, 0, stream>>>(wheads, Vb);
  prep_rt<<<G((long)NHEAD * KLEN * DHEAD),       256, 0, stream>>>(rheadk, Rt);

  // ---- 5. AC / BD : per (b,n), M=256 N=512 K=64 ----
  gemmL(stream, Aq, Kt, ACb, nullptr, nullptr, nullptr,
        QLEN, KLEN, DHEAD, DHEAD, DHEAD, KLEN,
        (long)QLEN * DHEAD, 0, 1, (long)KLEN * DHEAD, 0, 1,
        (long)QLEN * KLEN, 0, 1, 0, 0, BSZ * NHEAD);
  gemmL(stream, Aqr, Rt, BDb, nullptr, nullptr, nullptr,
        QLEN, KLEN, DHEAD, DHEAD, DHEAD, KLEN,
        (long)QLEN * DHEAD, 0, 1, 0, (long)KLEN * DHEAD, NHEAD,
        (long)QLEN * KLEN, 0, 1, 0, 0, BSZ * NHEAD);

  // ---- 6. rel-shift + mask + softmax -> prob (bf16) ----
  softmax_kernel<<<BSZ * NHEAD * QLEN, 256, 0, stream>>>(ACb, BDb, prob);

  // ---- 7. attn_vec^T = prob @ v^T : per (b,n), M=256 N=64 K=512,
  //         scattered into avF/avB [b][i][n*64+d] ----
  gemmL(stream, prob, Vb, avF, nullptr, nullptr, avB,
        QLEN, DHEAD, KLEN, KLEN, KLEN, DMODEL,
        (long)QLEN * KLEN, 0, 1,
        (long)DMODEL * KLEN, (long)DHEAD * KLEN, NHEAD,
        (long)QLEN * DMODEL, DHEAD, NHEAD,
        0, EPI_BF16, BSZ * NHEAD);

  // ---- 8. attn_out = Wo @ attn_vec + bo : per b, M=1024 N=256 K=1024 ----
  gemmL(stream, WoB, avB, attnO, nullptr, bo, nullptr,
        DMODEL, QLEN, DMODEL, DMODEL, DMODEL, QLEN,
        0, 0, 1, (long)QLEN * DMODEL, 0, 1, (long)DMODEL * QLEN, 0, 1,
        0, EPI_BIASM, BSZ);

  // ---- 9. h = LN(attn_out + z1ss) ----
  ln_h_kernel<<<BSZ * QLEN, 256, 0, stream>>>(attnO, z1ss, hF, hB);

  // ---- 10. ff1 = relu(h @ W1^T + b1) : per b, M=256 N=4096 K=1024 ----
  gemmL(stream, hB, W1B, ff1F, nullptr, b1, ff1B,
        QLEN, DINNER, DMODEL, DMODEL, DMODEL, DINNER,
        (long)QLEN * DMODEL, 0, 1, 0, 0, 1, (long)QLEN * DINNER, 0, 1,
        0, EPI_BIASN | EPI_RELU | EPI_BF16, BSZ);

  // ---- 11. ff2 = ff1 @ W2^T + b2 : per b, M=256 N=1024 K=4096 ----
  gemmL(stream, ff1B, W2B, ff2F, nullptr, b2, nullptr,
        QLEN, DMODEL, DINNER, DINNER, DINNER, DMODEL,
        (long)QLEN * DINNER, 0, 1, 0, 0, 1, (long)QLEN * DMODEL, 0, 1,
        0, EPI_BIASN, BSZ);

  // ---- 12. out = LN(ff2 + h), transposed to [b, d, q] ----
  ln_out_kernel<<<BSZ * QLEN, 256, 0, stream>>>(ff2F, hF, out);
}
